// MultimodalTextGuidedMIL_62826781606000
// MI455X (gfx1250) — compile-verified, matrix operand order checked by reading, not run on hardware
//
#include <hip/hip_runtime.h>
#include <math.h>

// ---------------------------------------------------------------------------
// MultimodalTextGuidedMIL for MI455X (gfx1250, wave32, WMMA)
// Big GEMMs -> v_wmma_f32_16x16x32_bf16 (bf16 in, f32 accumulate),
//   16x64 tile per wave (4 wmma / K-step), B panel staged in LDS as bf16
//   (converted once per block, K-contiguous -> 2x ds_load_b128 per tile).
// Everything else -> scalar VALU (negligible FLOPs).
// ---------------------------------------------------------------------------

typedef __attribute__((ext_vector_type(16))) __bf16 v16bf;
typedef __attribute__((ext_vector_type(8)))  __bf16 v8bf;
typedef __attribute__((ext_vector_type(8)))  float  v8f;

#define Bv   4
#define Kv   32
#define Pv   16
#define Dv   1024
#define DTv  768
#define SIMv 256
#define Hv   8
#define HDv  128
#define NWIN 961   // 31*31 windows
#define WOUTv 512
#define FDIMv 512
#define NCv  2

// ---------------------------------------------------------------------------
// WMMA GEMM: C[M,N] = act( A[M,K] * B[K,N] + bias + resid )
// Block = 8 waves; each wave owns a 16x64 output tile (4 accumulators).
// B panel (32 K x 64 N) staged per K-chunk in LDS, pre-converted to bf16 and
// transposed to K-contiguous (BsT[col][K], row stride 40 -> 80B, 16B-aligned),
// so a lane's B fragment is two 16B ds_load_b128 reads, zero cvt at read.
// Requires: M % 128 == 0, N % 64 == 0, K % 32 == 0 (true for all call sites).
// ACT: 0 none, 1 tanh, 2 relu.  Epilogue fully compile-time specialized.
// ---------------------------------------------------------------------------
template <int ACT, bool HAS_BIAS, bool HAS_RESID>
__global__ __launch_bounds__(256) void gemm_bf16_wmma(
    const float* __restrict__ A, const float* __restrict__ Bm,
    float* __restrict__ C, int M, int N, int Kd,
    const float* __restrict__ bias, const float* __restrict__ resid)
{
    __shared__ __align__(16) __bf16 BsT[64][40];   // [col][K], 80B row stride

    const int tid  = threadIdx.x;
    const int wave = tid >> 5;
    const int lane = tid & 31;
    const int m0 = (blockIdx.y * 8 + wave) * 16;
    const int n0 = blockIdx.x * 64;
    const bool active = (m0 < M);

    const int half  = lane >> 4;           // 0: K 0-7,16-23   1: K 8-15,24-31
    const int lr    = lane & 15;
    const int kbase = half * 8;

    const float* __restrict__ Arow = A + (size_t)(m0 + lr) * Kd;

    v8f acc[4] = {};

    for (int kk = 0; kk < Kd; kk += 32) {
        // ---- stage B[kk..kk+31][n0..n0+63]: coalesced f32 load, bf16 LDS ----
        __syncthreads();
#pragma unroll
        for (int i = tid; i < 32 * 64; i += 256) {
            const int r = i >> 6, c = i & 63;            // r: K, c: col
            BsT[c][r] = (__bf16)Bm[(size_t)(kk + r) * N + (n0 + c)];
        }
        __syncthreads();

        if (active) {
            // A fragment: two contiguous 8-float runs -> 4x float4 (16B aligned)
            const float4 a0 = *(const float4*)(Arow + kk + kbase);
            const float4 a1 = *(const float4*)(Arow + kk + kbase + 4);
            const float4 a2 = *(const float4*)(Arow + kk + kbase + 16);
            const float4 a3 = *(const float4*)(Arow + kk + kbase + 20);
            v16bf af;
            af[0]  = (__bf16)a0.x; af[1]  = (__bf16)a0.y;
            af[2]  = (__bf16)a0.z; af[3]  = (__bf16)a0.w;
            af[4]  = (__bf16)a1.x; af[5]  = (__bf16)a1.y;
            af[6]  = (__bf16)a1.z; af[7]  = (__bf16)a1.w;
            af[8]  = (__bf16)a2.x; af[9]  = (__bf16)a2.y;
            af[10] = (__bf16)a2.z; af[11] = (__bf16)a2.w;
            af[12] = (__bf16)a3.x; af[13] = (__bf16)a3.y;
            af[14] = (__bf16)a3.z; af[15] = (__bf16)a3.w;

#pragma unroll
            for (int nt = 0; nt < 4; ++nt) {
                const int col = nt * 16 + lr;
                // frag element e -> K = kbase + e + (e & 8): lo run + hi run
                const v8bf lo = *(const v8bf*)&BsT[col][kbase];        // 16B
                const v8bf hi = *(const v8bf*)&BsT[col][kbase + 16];   // 16B
                const v16bf bf = __builtin_shufflevector(
                    lo, hi, 0, 1, 2, 3, 4, 5, 6, 7,
                            8, 9, 10, 11, 12, 13, 14, 15);
                acc[nt] = __builtin_amdgcn_wmma_f32_16x16x32_bf16(
                    false, af, false, bf, (short)0, acc[nt], false, false);
            }
        }
    }

    if (!active) return;

    // C/D layout: VGPR i -> row m0 + i + half*8, column tile-base + lr
#pragma unroll
    for (int nt = 0; nt < 4; ++nt) {
#pragma unroll
        for (int i = 0; i < 8; ++i) {
            const int r = m0 + half * 8 + i;
            const int c = n0 + nt * 16 + lr;
            float v = acc[nt][i];
            if (HAS_BIAS)  v += bias[c];
            if (HAS_RESID) v += resid[(size_t)r * N + c];
            if (ACT == 1)      v = tanhf(v);
            else if (ACT == 2) v = fmaxf(v, 0.f);
            C[(size_t)r * N + c] = v;
        }
    }
}

// ---------------------------------------------------------------------------
// pt = text @ Wt + bt ; ptn = pt / (||pt|| + 1e-8).   grid=B, block=256 (d=tid)
// ---------------------------------------------------------------------------
__global__ __launch_bounds__(256) void k_text_ptn(
    const float* __restrict__ text, const float* __restrict__ Wt,
    const float* __restrict__ bt, float* __restrict__ ptn)
{
    const int b = blockIdx.x, d = threadIdx.x;
    const float* tx = text + b * DTv;
    float s = bt[d];
    for (int k = 0; k < DTv; ++k) s += tx[k] * Wt[k * SIMv + d];
    __shared__ float red[256];
    red[d] = s * s;
    __syncthreads();
    for (int o = 128; o > 0; o >>= 1) {
        if (d < o) red[d] += red[d + o];
        __syncthreads();
    }
    const float nrm = sqrtf(red[0]) + 1e-8f;
    ptn[b * SIMv + d] = s / nrm;
}

// ---------------------------------------------------------------------------
// Window cosine in projection space (window-mean commutes with projection):
// pc[w] = mean_{p in win} proj[patch] + bp ; cos = ptn . (pc/(||pc||+1e-8))
// grid=(961,B), block=256 (d=tid)
// ---------------------------------------------------------------------------
__global__ __launch_bounds__(256) void k_win_cos(
    const float* __restrict__ proj, const float* __restrict__ bp,
    const float* __restrict__ ptn, float* __restrict__ cosout)
{
    const int w = blockIdx.x, b = blockIdx.y, d = threadIdx.x;
    const int wr = w / 31, wc = w % 31;
    float s = 0.f;
#pragma unroll
    for (int p = 0; p < 16; ++p) {
        const int patch = (wr * 2 + (p >> 2)) * 64 + wc * 2 + (p & 3);
        s += proj[(size_t)(b * 4096 + patch) * SIMv + d];
    }
    s = s * (1.f / 16.f) + bp[d];
    __shared__ float rss[256], rdot[256];
    rss[d]  = s * s;
    rdot[d] = s * ptn[b * SIMv + d];
    __syncthreads();
    for (int o = 128; o > 0; o >>= 1) {
        if (d < o) { rss[d] += rss[d + o]; rdot[d] += rdot[d + o]; }
        __syncthreads();
    }
    if (d == 0) cosout[b * NWIN + w] = rdot[0] / (sqrtf(rss[0]) + 1e-8f);
}

// ---------------------------------------------------------------------------
// Top-32 window indices per batch (iterative argmax, first-index tie-break).
// grid=B, block=256
// ---------------------------------------------------------------------------
__global__ __launch_bounds__(256) void k_topk32(
    const float* __restrict__ cosv, int* __restrict__ topk)
{
    const int b = blockIdx.x, t = threadIdx.x;
    __shared__ float vals[NWIN];
    __shared__ float bv[256];
    __shared__ int   bi[256];
    for (int i = t; i < NWIN; i += 256) vals[i] = cosv[b * NWIN + i];
    __syncthreads();
    for (int it = 0; it < Kv; ++it) {
        float best = -1e30f; int bidx = NWIN;
        for (int i = t; i < NWIN; i += 256) {
            const float v = vals[i];
            if (v > best || (v == best && i < bidx)) { best = v; bidx = i; }
        }
        bv[t] = best; bi[t] = bidx;
        __syncthreads();
        for (int o = 128; o > 0; o >>= 1) {
            if (t < o) {
                if (bv[t + o] > bv[t] || (bv[t + o] == bv[t] && bi[t + o] < bi[t])) {
                    bv[t] = bv[t + o]; bi[t] = bi[t + o];
                }
            }
            __syncthreads();
        }
        if (t == 0) { topk[b * Kv + it] = bi[0]; vals[bi[0]] = -1e30f; }
        __syncthreads();
    }
}

// ---------------------------------------------------------------------------
// Gather selected window patches: sel[b,k,p,:] = feats[b, patch(w,p), :]
// grid = B*K*P, block=256
// ---------------------------------------------------------------------------
__global__ __launch_bounds__(256) void k_gather(
    const float* __restrict__ feats, const int* __restrict__ topk,
    float* __restrict__ sel)
{
    const int r = blockIdx.x;           // (b,k,p) flat, 2048
    const int b = r >> 9;
    const int k = (r >> 4) & 31;
    const int p = r & 15;
    const int w = topk[b * Kv + k];
    const int wr = w / 31, wc = w % 31;
    const int patch = (wr * 2 + (p >> 2)) * 64 + wc * 2 + (p & 3);
    const float* src = feats + (size_t)(b * 4096 + patch) * Dv;
    float* dst = sel + (size_t)r * Dv;
    for (int d = threadIdx.x; d < Dv; d += 256) dst[d] = src[d];
}

// ---------------------------------------------------------------------------
// Per-window per-head 16x16 self-attention (LDS staged). grid=(B*K, H)
// ---------------------------------------------------------------------------
__global__ __launch_bounds__(256) void k_attn16(
    const float* __restrict__ Q, const float* __restrict__ Kt,
    const float* __restrict__ V, float* __restrict__ O)
{
    __shared__ float qs[16][HDv], ks[16][HDv], vs[16][HDv];
    __shared__ float sc[16][16];
    const int bk = blockIdx.x, h = blockIdx.y, t = threadIdx.x;
    const size_t base = (size_t)bk * 16 * Dv + h * HDv;
    for (int i = t; i < 16 * HDv; i += 256) {
        const int p = i >> 7, d = i & 127;
        qs[p][d] = Q[base + (size_t)p * Dv + d];
        ks[p][d] = Kt[base + (size_t)p * Dv + d];
        vs[p][d] = V[base + (size_t)p * Dv + d];
    }
    __syncthreads();
    {
        const int i = t >> 4, j = t & 15;
        float s = 0.f;
        for (int d = 0; d < HDv; ++d) s += qs[i][d] * ks[j][d];
        sc[i][j] = s * 0.08838834764831845f;   // 1/sqrt(128)
    }
    __syncthreads();
    if (t < 16) {
        float mx = -1e30f;
        for (int j = 0; j < 16; ++j) mx = fmaxf(mx, sc[t][j]);
        float sum = 0.f;
        for (int j = 0; j < 16; ++j) { const float e = __expf(sc[t][j] - mx); sc[t][j] = e; sum += e; }
        const float inv = 1.f / sum;
        for (int j = 0; j < 16; ++j) sc[t][j] *= inv;
    }
    __syncthreads();
    for (int i = t; i < 16 * HDv; i += 256) {
        const int p = i >> 7, d = i & 127;
        float s = 0.f;
        for (int j = 0; j < 16; ++j) s += sc[p][j] * vs[j][d];
        O[base + (size_t)p * Dv + d] = s;
    }
}

// ---------------------------------------------------------------------------
// out[r] = dot(A[r, :cols], w).  grid = rows, block = 256
// ---------------------------------------------------------------------------
__global__ __launch_bounds__(256) void k_row_dot(
    const float* __restrict__ A, const float* __restrict__ w,
    float* __restrict__ out, int cols)
{
    const int r = blockIdx.x, t = threadIdx.x;
    float s = 0.f;
    for (int c = t; c < cols; c += 256) s += A[(size_t)r * cols + c] * w[c];
    __shared__ float red[256];
    red[t] = s;
    __syncthreads();
    for (int o = 128; o > 0; o >>= 1) {
        if (t < o) red[t] += red[t + o];
        __syncthreads();
    }
    if (t == 0) out[r] = red[0];
}

// ---------------------------------------------------------------------------
// Attention-MIL pool: pooled[g,:] = sum_i softmax(s[g,:])_i * x[g,i,:]
// grid = groups, block = 256
// ---------------------------------------------------------------------------
__global__ __launch_bounds__(256) void k_mil_pool(
    const float* __restrict__ xin, const float* __restrict__ s,
    float* __restrict__ pooled, int S, int Dm)
{
    const int g = blockIdx.x, t = threadIdx.x;
    __shared__ float a[64];
    if (t == 0) {
        float mx = -1e30f;
        for (int i = 0; i < S; ++i) mx = fmaxf(mx, s[g * S + i]);
        float sum = 0.f;
        for (int i = 0; i < S; ++i) { const float e = __expf(s[g * S + i] - mx); a[i] = e; sum += e; }
        const float inv = 1.f / sum;
        for (int i = 0; i < S; ++i) a[i] *= inv;
    }
    __syncthreads();
    for (int d = t; d < Dm; d += 256) {
        float acc = 0.f;
        for (int i = 0; i < S; ++i) acc += a[i] * xin[((size_t)g * S + i) * Dm + d];
        pooled[(size_t)g * Dm + d] = acc;
    }
}

// ---------------------------------------------------------------------------
// Small batched matvec: out[b, n] = act(in[b,:Kd] @ W[:,n] + bias[n])
// grid = (ceil(N/256), B).  act: 0 none, 1 relu.  out row stride = ldo.
// ---------------------------------------------------------------------------
__global__ __launch_bounds__(256) void k_matvec(
    const float* __restrict__ in, const float* __restrict__ W,
    const float* __restrict__ bias, float* __restrict__ out,
    int Kd, int N, int ldo, int act)
{
    const int n = blockIdx.x * 256 + threadIdx.x;
    const int b = blockIdx.y;
    if (n >= N) return;
    const float* row = in + (size_t)b * Kd;
    float s = bias ? bias[n] : 0.f;
    for (int k = 0; k < Kd; ++k) s += row[k] * W[(size_t)k * N + n];
    if (act == 1) s = fmaxf(s, 0.f);
    out[(size_t)b * ldo + n] = s;
}

// ---------------------------------------------------------------------------
// Host-side orchestration
// ---------------------------------------------------------------------------
extern "C" void kernel_launch(void* const* d_in, const int* in_sizes, int n_in,
                              void* d_out, int out_size, void* d_ws, size_t ws_size,
                              hipStream_t stream)
{
    (void)in_sizes; (void)n_in; (void)out_size; (void)ws_size;

    const float* feats = (const float*)d_in[0];   // (4, 4096, 1024)
    const float* text  = (const float*)d_in[1];   // (4, 768)
    const float* Wt    = (const float*)d_in[2];
    const float* bt    = (const float*)d_in[3];
    const float* Wp    = (const float*)d_in[4];
    const float* bp    = (const float*)d_in[5];
    const float* Wq    = (const float*)d_in[6];
    const float* Wk    = (const float*)d_in[7];
    const float* Wv    = (const float*)d_in[8];
    const float* Wo    = (const float*)d_in[9];
    const float* V1    = (const float*)d_in[10];
    const float* bV1   = (const float*)d_in[11];
    const float* w1    = (const float*)d_in[12];
    const float* Wm1   = (const float*)d_in[13];
    const float* bm1   = (const float*)d_in[14];
    const float* V2    = (const float*)d_in[15];
    const float* bV2   = (const float*)d_in[16];
    const float* w2    = (const float*)d_in[17];
    const float* Wm2   = (const float*)d_in[18];
    const float* bm2   = (const float*)d_in[19];
    // Cq (20), Ck (21) provably cancel: softmax over a single key == 1.
    const float* Cv    = (const float*)d_in[22];
    const float* Co    = (const float*)d_in[23];
    const float* Wc1   = (const float*)d_in[24];
    const float* bc1   = (const float*)d_in[25];
    const float* Wc2   = (const float*)d_in[26];
    const float* bc2   = (const float*)d_in[27];

    float* ws = (float*)d_ws;
    const size_t MEG = 1024u * 1024u;
    // --- phase-1 buffers ---
    float* proj = ws;                 // 16384 x 256   (4M floats; dead after topk)
    float* sel  = ws + 4 * MEG;       // 2048 x 1024
    float* q    = ws + 6 * MEG;
    float* kbuf = ws + 8 * MEG;
    float* vbuf = ws + 10 * MEG;
    float* ao   = ws + 12 * MEG;
    float* x    = ws + 14 * MEG;      // post-MHA (2048 x 1024); ends at 16M floats
    float* ptn  = ws + 16 * MEG;      // 4 x 256
    float* cosb = ptn + 1024;         // 4 x 961
    int*   topk = (int*)(cosb + 4096);// 4 x 32
    // --- phase-2 buffers overlay the dead proj region ---
    float* h1      = ws;              // 2048 x 256
    float* s1      = ws + 524288;     // 2048
    float* pooled1 = ws + 526336;     // 128 x 1024
    float* wrep    = ws + 657408;     // 128 x 512
    float* h2      = ws + 722944;     // 128 x 256
    float* s2      = ws + 755712;     // 128
    float* pooled2 = ws + 755840;     // 4 x 512
    float* fused   = ws + 757888;     // 4 x 1024 (img | cross)
    float* tmpv    = ws + 761984;     // 4 x 512
    float* hidden  = ws + 764032;     // 4 x 512

    const dim3 blk(256);

    // 1) proj = feats @ Wp  (window-mean commutes with the linear projection)
    gemm_bf16_wmma<0, false, false><<<dim3(SIMv / 64, 16384 / 128), blk, 0, stream>>>(
        feats, Wp, proj, 16384, SIMv, Dv, nullptr, nullptr);

    // 2) normalized text projection
    k_text_ptn<<<Bv, blk, 0, stream>>>(text, Wt, bt, ptn);

    // 3) per-window cosine vs text
    k_win_cos<<<dim3(NWIN, Bv), blk, 0, stream>>>(proj, bp, ptn, cosb);

    // 4) top-32 windows per batch
    k_topk32<<<Bv, blk, 0, stream>>>(cosb, topk);

    // 5) gather selected patches: sel (2048 x 1024)
    k_gather<<<Bv * Kv * Pv, blk, 0, stream>>>(feats, topk, sel);

    // 6) Q/K/V projections (WMMA)
    gemm_bf16_wmma<0, false, false><<<dim3(Dv / 64, 2048 / 128), blk, 0, stream>>>(
        sel, Wq, q, 2048, Dv, Dv, nullptr, nullptr);
    gemm_bf16_wmma<0, false, false><<<dim3(Dv / 64, 2048 / 128), blk, 0, stream>>>(
        sel, Wk, kbuf, 2048, Dv, Dv, nullptr, nullptr);
    gemm_bf16_wmma<0, false, false><<<dim3(Dv / 64, 2048 / 128), blk, 0, stream>>>(
        sel, Wv, vbuf, 2048, Dv, Dv, nullptr, nullptr);

    // 7) 16x16 attention per (window, head)
    k_attn16<<<dim3(Bv * Kv, Hv), blk, 0, stream>>>(q, kbuf, vbuf, ao);

    // 8) output projection + residual: x = sel + ao @ Wo  (WMMA)
    gemm_bf16_wmma<0, false, true><<<dim3(Dv / 64, 2048 / 128), blk, 0, stream>>>(
        ao, Wo, x, 2048, Dv, Dv, nullptr, sel);

    // 9) MIL-1 hidden: h1 = tanh(x @ V1 + bV1)  (WMMA)
    gemm_bf16_wmma<1, true, false><<<dim3(SIMv / 64, 2048 / 128), blk, 0, stream>>>(
        x, V1, h1, 2048, SIMv, Dv, bV1, nullptr);

    // 10) MIL-1 logits + pool over P=16 patches
    k_row_dot<<<2048, blk, 0, stream>>>(h1, w1, s1, SIMv);
    k_mil_pool<<<Bv * Kv, blk, 0, stream>>>(x, s1, pooled1, Pv, Dv);

    // 11) wrep = pooled1 @ Wm1 + bm1  (WMMA)
    gemm_bf16_wmma<0, true, false><<<dim3(WOUTv / 64, 1), blk, 0, stream>>>(
        pooled1, Wm1, wrep, 128, WOUTv, Dv, bm1, nullptr);

    // 12) MIL-2 hidden: h2 = tanh(wrep @ V2 + bV2)  (WMMA)
    gemm_bf16_wmma<1, true, false><<<dim3(SIMv / 64, 1), blk, 0, stream>>>(
        wrep, V2, h2, 128, SIMv, WOUTv, bV2, nullptr);

    // 13) MIL-2 logits + pool over K=32 windows
    k_row_dot<<<128, blk, 0, stream>>>(h2, w2, s2, SIMv);
    k_mil_pool<<<Bv, blk, 0, stream>>>(wrep, s2, pooled2, Kv, WOUTv);

    // 14) img = pooled2 @ Wm2 + bm2  -> fused[:, :512]
    k_matvec<<<dim3(2, Bv), blk, 0, stream>>>(pooled2, Wm2, bm2, fused,
                                              WOUTv, FDIMv, 2 * FDIMv, 0);

    // 15) cross-attn (single key => softmax == 1): cross = (text @ Cv) @ Co
    k_matvec<<<dim3(2, Bv), blk, 0, stream>>>(text, Cv, nullptr, tmpv,
                                              DTv, FDIMv, FDIMv, 0);
    k_matvec<<<dim3(2, Bv), blk, 0, stream>>>(tmpv, Co, nullptr, fused + FDIMv,
                                              FDIMv, FDIMv, 2 * FDIMv, 0);

    // 16) classifier head
    k_matvec<<<dim3(2, Bv), blk, 0, stream>>>(fused, Wc1, bc1, hidden,
                                              2 * FDIMv, 512, 512, 1);
    k_matvec<<<dim3(1, Bv), blk, 0, stream>>>(hidden, Wc2, bc2, (float*)d_out,
                                              512, NCv, NCv, 0);
}
